// RungeKuttaIntegratorCell_grad_15599321219534
// MI455X (gfx1250) — matrix operand order repaired and checked by reading, not compiled, not run on hardware
//
#include <hip/hip_runtime.h>

typedef __attribute__((ext_vector_type(16))) _Float16 v16h;
typedef __attribute__((ext_vector_type(8)))  float    v8f;

#define BLOCK 256
#define GRID_CAP 2048

// RBF(y) = sum_j w[j] * exp(-g[j]*(y-c[j])^2), 64 centers.
// cg: interleaved (center, -gamma*log2(e)) pairs in LDS.
// a0/a1: f16 weight broadcast in WMMA A-layout (K=0..31 / K=32..63).
// kbase = (lane>>4)*16 : this lane's K-range inside each 16x16x32 WMMA.
__device__ __forceinline__ float rbf_eval(float y, v16h a0, v16h a1, int kbase,
                                          const float2* __restrict__ cg) {
    v16h b;
    v8f acc = {};
    #pragma unroll
    for (int e = 0; e < 16; ++e) {
        float2 p = cg[kbase + e];                 // p.x = c, p.y = -g*log2e
        float d = y - p.x;
        b[e] = (_Float16)__builtin_amdgcn_exp2f(p.y * d * d);
    }
    acc = __builtin_amdgcn_wmma_f32_16x16x32_f16(false, a0, false, b,
                                                 (short)0, acc, false, false);
    #pragma unroll
    for (int e = 0; e < 16; ++e) {
        float2 p = cg[kbase + 32 + e];
        float d = y - p.x;
        b[e] = (_Float16)__builtin_amdgcn_exp2f(p.y * d * d);
    }
    acc = __builtin_amdgcn_wmma_f32_16x16x32_f16(false, a1, false, b,
                                                 (short)0, acc, false, false);
    return acc[0];   // all 16x16 outputs in column n hold r_n -> every lane has its value
}

__global__ void __launch_bounds__(BLOCK)
rk4_rbf_kernel(const float* __restrict__ u,
               const float* __restrict__ states,   // [2,B]; only row 1 is read
               const float* __restrict__ centers,
               const float* __restrict__ gammas,
               const float* __restrict__ weights,
               float* __restrict__ out,            // [2,B]
               int B) {
    __shared__ float s_cg[128];   // interleaved (c, -g*log2e)
    __shared__ float s_w[64];

    const int tid = threadIdx.x;
    if (tid < 64) {
        s_cg[2 * tid]     = centers[tid];
        s_cg[2 * tid + 1] = gammas[tid] * (-1.4426950408889634f); // fold log2e + negate
        s_w[tid]          = weights[tid];
    }
    __syncthreads();

    const float2* cg = (const float2*)s_cg;
    const int lane = tid & 31;
    const int n    = lane & 15;       // element slot within wave
    const int hi   = lane >> 4;       // half-wave id (K split)
    const int wave = blockIdx.x * (BLOCK / 32) + (tid >> 5);
    const int nwaves = gridDim.x * (BLOCK / 32);
    const int stride = nwaves * 16;

    // Build A = broadcast weights in contiguous K order (B uses the same):
    // lane (hi half), vector element e  <->  K = e + hi*16
    v16h a0, a1;
    #pragma unroll
    for (int e = 0; e < 16; ++e) {
        const int k = e + hi * 16;
        a0[e] = (_Float16)s_w[k];
        a1[e] = (_Float16)s_w[k + 32];
    }
    const int kbase = hi * 16;

    const float KFC = 214.9261f + 19.3607f;   // F_V + F_C
    const float OFF = -3.2902f;
    const float MM  = 1.0f / 95.45f;
    const float DTv = 0.005f;
    const float DT6 = DTv / 6.0f;

    const float* __restrict__ y1p = states + B;   // row 1

    for (int base = wave * 16; base < B; base += stride) {
        const int i  = base + n;
        const int ic = (i < B) ? i : (B - 1);     // clamp: keep EXEC uniform for WMMA

        // speculative prefetch of next chunk (silently dropped if OOB)
        __builtin_prefetch(y1p + ic + stride, 0, 0);
        __builtin_prefetch(u   + ic + stride, 0, 0);

        const float y  = y1p[ic];
        const float uu = u[ic];

        // k1 (== yddoti since A_COEF[0]=0)
        const float r0 = rbf_eval(y, a0, a1, kbase, cg);
        const float f  = (uu - KFC * y - OFF - r0) * MM;

        const float y2 = fmaf(0.5f * DTv, f, y);  // stage 2 == stage 3
        const float y4 = fmaf(DTv, f, y);

        const float r2 = rbf_eval(y2, a0, a1, kbase, cg);
        const float r4 = rbf_eval(y4, a0, a1, kbase, cg);

        const float k2 = (uu - KFC * y2 - OFF - r2) * MM;
        const float k4 = (uu - KFC * y4 - OFF - r4) * MM;

        const float row1 = DT6 * (f + 4.0f * k2 + k4);            // out[1][i]
        const float row0 = -(DT6 * MM) * (y + 4.0f * y2 + y4);    // out[0][i]

        // lanes 0-15 store row1, lanes 16-31 store row0 (full-wave store)
        const float val = hi ? row0 : row1;
        float* addr = out + i + (hi ? 0 : B);
        if (i < B) *addr = val;
    }
}

extern "C" void kernel_launch(void* const* d_in, const int* in_sizes, int n_in,
                              void* d_out, int out_size, void* d_ws, size_t ws_size,
                              hipStream_t stream) {
    const float* u       = (const float*)d_in[0];
    const float* states  = (const float*)d_in[1];
    const float* centers = (const float*)d_in[2];
    const float* gammas  = (const float*)d_in[3];
    const float* weights = (const float*)d_in[4];
    float* out = (float*)d_out;
    const int B = in_sizes[0];

    int grid = (B + (BLOCK / 32) * 16 - 1) / ((BLOCK / 32) * 16);
    if (grid > GRID_CAP) grid = GRID_CAP;
    if (grid < 1) grid = 1;

    rk4_rbf_kernel<<<grid, BLOCK, 0, stream>>>(u, states, centers, gammas, weights, out, B);
}
